// Head_14010183320284
// MI455X (gfx1250) — matrix operand order repaired and checked by reading, not compiled, not run on hardware
//
#include <hip/hip_runtime.h>

typedef __attribute__((ext_vector_type(16))) _Float16 v16h;
typedef __attribute__((ext_vector_type(8)))  _Float16 v8h;
typedef __attribute__((ext_vector_type(8)))  float    v8f;
typedef __attribute__((ext_vector_type(4)))  unsigned int u32x4;
typedef __attribute__((ext_vector_type(8)))  int      i32x8;
typedef __attribute__((ext_vector_type(4)))  int      i32x4;

#define HS    128
#define CEMB  2048
#define TSEQ  4096
#define BATCH 4

#if defined(__has_builtin)
#  if __has_builtin(__builtin_amdgcn_tensor_load_to_lds)
#    define HAVE_TDM 1
#  else
#    define HAVE_TDM 0
#  endif
#else
#  define HAVE_TDM 0
#endif

static __device__ __forceinline__ v16h cat8(v8h lo, v8h hi) {
  return __builtin_shufflevector(lo, hi, 0,1,2,3,4,5,6,7,8,9,10,11,12,13,14,15);
}

// ---------------------------------------------------------------------------
// Stage 1: fused QKV projection GEMM, f16 WMMA, f32 accumulate.
// grid = (128, 3); block tile 128(M) x 128(N); K-chunks of 32, reg-prefetched.
// ---------------------------------------------------------------------------
#define S1_PAD 56   // f16 per LDS row (32 data + pad) = 112B, 16B aligned

__global__ __launch_bounds__(256)
void proj_kernel(const float* __restrict__ x,
                 const float* __restrict__ Wq,
                 const float* __restrict__ Wk,
                 const float* __restrict__ Wv,
                 _Float16* __restrict__ q,
                 _Float16* __restrict__ k,
                 _Float16* __restrict__ v)
{
  __shared__ _Float16 a_lds[128 * S1_PAD];   // x tile   [m][k]
  __shared__ _Float16 bT_lds[128 * S1_PAD];  // W tile T [n][k]

  const float* W = (blockIdx.y == 0) ? Wq : (blockIdx.y == 1) ? Wk : Wv;
  _Float16*  out = (blockIdx.y == 0) ? q  : (blockIdx.y == 1) ? k  : v;

  const int tid  = threadIdx.x;
  const int wave = tid >> 5;
  const int lane = tid & 31;
  const int half = lane >> 4;
  const int l16  = lane & 15;
  const int m0   = blockIdx.x * 128;

  v8f acc[8];
  #pragma unroll
  for (int i = 0; i < 8; ++i) acc[i] = (v8f){};

  float4 xa[4], wb[4];
  auto loadG = [&](int k0) {
    #pragma unroll
    for (int it = 0; it < 4; ++it) {
      int i = tid + it * 256;                       // float4 index 0..1023
      xa[it] = *(const float4*)(x + (size_t)(m0 + (i >> 3)) * CEMB + k0 + (i & 7) * 4);
      wb[it] = *(const float4*)(W + (size_t)(k0 + (i >> 5)) * HS + (i & 31) * 4);
    }
  };

  loadG(0);
  for (int k0 = 0; k0 < CEMB; k0 += 32) {
    __syncthreads();
    #pragma unroll
    for (int it = 0; it < 4; ++it) {
      int i   = tid + it * 256;
      int row = i >> 3, c = (i & 7) * 4;
      _Float16* d = &a_lds[row * S1_PAD + c];
      d[0] = (_Float16)xa[it].x; d[1] = (_Float16)xa[it].y;
      d[2] = (_Float16)xa[it].z; d[3] = (_Float16)xa[it].w;
      int kk = i >> 5, n = (i & 31) * 4;
      bT_lds[(n + 0) * S1_PAD + kk] = (_Float16)wb[it].x;
      bT_lds[(n + 1) * S1_PAD + kk] = (_Float16)wb[it].y;
      bT_lds[(n + 2) * S1_PAD + kk] = (_Float16)wb[it].z;
      bT_lds[(n + 3) * S1_PAD + kk] = (_Float16)wb[it].w;
    }
    __syncthreads();
    if (k0 + 32 < CEMB) loadG(k0 + 32);            // overlap with WMMAs below

    // A frag: lane(l16)=row M; elems 0-7: K=half*8+e, 8-15: K=16+half*8+e
    const _Float16* arow = &a_lds[(wave * 16 + l16) * S1_PAD];
    v16h afrag = cat8(*(const v8h*)(arow + half * 8),
                      *(const v8h*)(arow + 16 + half * 8));

    // Preload ALL 8 B fragments (16 independent ds_load_b128) so the WMMA
    // stream issues back-to-back with graduated s_wait_dscnt, not 0x0 stalls.
    v16h bfr[8];
    #pragma unroll
    for (int nt = 0; nt < 8; ++nt) {
      const _Float16* brow = &bT_lds[(nt * 16 + l16) * S1_PAD];
      bfr[nt] = cat8(*(const v8h*)(brow + half * 8),
                     *(const v8h*)(brow + 16 + half * 8));
    }
    #pragma unroll
    for (int nt = 0; nt < 8; ++nt)
      acc[nt] = __builtin_amdgcn_wmma_f32_16x16x32_f16(
          false, afrag, false, bfr[nt], (short)0, acc[nt], false, false);
  }

  #pragma unroll
  for (int nt = 0; nt < 8; ++nt)
    #pragma unroll
    for (int r = 0; r < 8; ++r) {
      int row = m0 + wave * 16 + r + 8 * half;
      out[(size_t)row * HS + nt * 16 + l16] = (_Float16)acc[nt][r];
    }
}

// ---------------------------------------------------------------------------
// Stage 2: causal flash attention; K tile via TDM (tensor_load_to_lds) with
// double-buffered LDS; V manually transposed with register prefetch.
// ---------------------------------------------------------------------------
#define KV    64
#define KPAD  136   // 128 + 8 f16 -> 272B pitch (TDM pad produces this)
#define VPAD  72    // 64 + 8 f16 -> 144B pitch
#define PPAD  56

__global__ __launch_bounds__(256)
void attn_kernel(const _Float16* __restrict__ Q,
                 const _Float16* __restrict__ K,
                 const _Float16* __restrict__ V,
                 float* __restrict__ out)
{
  __shared__ _Float16 k_lds[2][KV * KPAD];     // [key][d]
  __shared__ _Float16 vt_lds[2][HS * VPAD];    // [d][key] (transposed)
  __shared__ _Float16 p_lds[8][16 * PPAD];     // per-wave P re-layout pad

  const int tid  = threadIdx.x;
  const int wave = tid >> 5;
  const int lane = tid & 31;
  const int half = lane >> 4;
  const int l16  = lane & 15;
  const int qb   = (int)gridDim.x - 1 - (int)blockIdx.x;  // heavy blocks first
  const size_t boff = (size_t)blockIdx.y * TSEQ * HS;
  const int qrow0 = qb * 128 + wave * 16;

  // Q fragments (4 D-chunks of 32) straight from global row-major f16
  v16h qfrag[4];
  {
    const _Float16* qrow = Q + boff + (size_t)(qrow0 + l16) * HS;
    #pragma unroll
    for (int c = 0; c < 4; ++c)
      qfrag[c] = cat8(*(const v8h*)(qrow + c * 32 + half * 8),
                      *(const v8h*)(qrow + c * 32 + 16 + half * 8));
  }

  v8f accO[8];
  #pragma unroll
  for (int i = 0; i < 8; ++i) accO[i] = (v8f){};
  float mrow[8], lrow[8];
  #pragma unroll
  for (int r = 0; r < 8; ++r) { mrow[r] = -1e30f; lrow[r] = 0.0f; }

  const float scale = 0.08838834764831845f;    // 1/sqrt(128)
  const int kc_end  = 2 * qb + 2;

  uint4 vreg[4];
#if !HAVE_TDM
  uint4 kreg[4];
#endif
  auto loadChunk = [&](int kc) {               // global -> registers
    const _Float16* vg = V + boff + (size_t)kc * KV * HS;
    #pragma unroll
    for (int it = 0; it < 4; ++it) {
      int i = tid + it * 256;
      vreg[it] = *(const uint4*)(vg + (size_t)(i >> 4) * HS + (i & 15) * 8);
    }
#if !HAVE_TDM
    const _Float16* kg = K + boff + (size_t)kc * KV * HS;
    #pragma unroll
    for (int it = 0; it < 4; ++it) {
      int i = tid + it * 256;
      kreg[it] = *(const uint4*)(kg + (size_t)(i >> 4) * HS + (i & 15) * 8);
    }
#endif
  };
  auto commitChunk = [&](int bufi) {           // registers -> LDS
    #pragma unroll
    for (int it = 0; it < 4; ++it) {
      int i = tid + it * 256;
      int row = i >> 4, c = (i & 15) * 8;
      const _Float16* h = (const _Float16*)&vreg[it];
      #pragma unroll
      for (int j = 0; j < 8; ++j) vt_lds[bufi][(c + j) * VPAD + row] = h[j];
#if !HAVE_TDM
      *(uint4*)&k_lds[bufi][row * KPAD + c] = kreg[it];
#endif
    }
  };

#if HAVE_TDM
  auto issueK = [&](int kc, int bufi) {        // TDM: K chunk -> LDS (padded)
    if (wave != 0) return;
    unsigned long long ga =
        (unsigned long long)(const void*)(K + boff + (size_t)kc * KV * HS);
    unsigned lds = (unsigned)(unsigned long long)(void*)&k_lds[bufi][0];
    u32x4 g0 = { 1u,                                   // count=1, user D#
                 lds,                                  // lds_addr
                 (unsigned)ga,
                 (unsigned)(ga >> 32) | (2u << 30) };  // addr hi | type=2
    int dim1 = TSEQ - kc * KV;                         // rows remaining
    i32x8 g1 = {
      (int)((1u << 16) | (1u << 20) | (5u << 22) | (3u << 25)),
      //     data=2B     pad_en       every 256B    pad 16B
      (int)(128u << 16),                 // tensor_dim0 = 128
      (int)((unsigned)dim1 << 16),       // tensor_dim1
      (int)(128u << 16),                 // tile_dim0 = 128
      64,                                // tile_dim1 = 64
      128,                               // tensor_dim0_stride = 128
      0, 0 };
    i32x4 z4 = {0, 0, 0, 0};
#if __clang_major__ >= 23
    i32x8 z8 = {0, 0, 0, 0, 0, 0, 0, 0};
    __builtin_amdgcn_tensor_load_to_lds(g0, g1, z4, z4, z8, 0);
#else
    __builtin_amdgcn_tensor_load_to_lds(g0, g1, z4, z4, 0);
#endif
  };
#endif

  // ---- prologue: chunk 0 ----
  loadChunk(0);
#if HAVE_TDM
  issueK(0, 0);
#endif
  commitChunk(0);
#if HAVE_TDM
  if (wave == 0) __builtin_amdgcn_s_wait_tensorcnt(0);
#endif
  __syncthreads();

  for (int kc = 0; kc < kc_end; ++kc) {
    const int buf = kc & 1;
    const bool pre = (kc + 1) < kc_end;
    if (pre) {
      loadChunk(kc + 1);                       // global latency hides in compute
#if HAVE_TDM
      issueK(kc + 1, buf ^ 1);
#endif
    }

    if (kc * KV <= qrow0 + 15) {               // wave-uniform causal trim
      const bool diag = (kc * KV + KV - 1) > qrow0;

      for (int ntp = 0; ntp < 2; ++ntp) {
        const int kbase = ntp * 32;
        if (kc * KV + kbase > qrow0 + 15) break;

        v8f s[2];
        #pragma unroll
        for (int sub = 0; sub < 2; ++sub) {
          v8f sa = (v8f){};
          const _Float16* krow = &k_lds[buf][(kbase + sub * 16 + l16) * KPAD];
          auto loadK = [&](int c) {
            return cat8(*(const v8h*)(krow + c * 32 + half * 8),
                        *(const v8h*)(krow + c * 32 + 16 + half * 8));
          };
          v16h kf = loadK(0);
          #pragma unroll
          for (int c = 0; c < 4; ++c) {
            v16h knx = (c < 3) ? loadK(c + 1) : kf;
            sa = __builtin_amdgcn_wmma_f32_16x16x32_f16(
                false, qfrag[c], false, kf, (short)0, sa, false, false);
            kf = knx;
          }
          s[sub] = sa;
        }

        #pragma unroll
        for (int sub = 0; sub < 2; ++sub)
          #pragma unroll
          for (int r = 0; r < 8; ++r) {
            float vv = s[sub][r] * scale;
            if (diag) {
              int kg = kc * KV + kbase + sub * 16 + l16;
              int qg = qrow0 + r + 8 * half;
              if (kg > qg) vv = -1e30f;
            }
            s[sub][r] = vv;
          }

        float corr[8];
        #pragma unroll
        for (int r = 0; r < 8; ++r) {
          float mx = fmaxf(s[0][r], s[1][r]);
          mx = fmaxf(mx, __shfl_xor(mx, 1));
          mx = fmaxf(mx, __shfl_xor(mx, 2));
          mx = fmaxf(mx, __shfl_xor(mx, 4));
          mx = fmaxf(mx, __shfl_xor(mx, 8));
          float mnew = fmaxf(mrow[r], mx);
          float p0 = __expf(s[0][r] - mnew);
          float p1 = __expf(s[1][r] - mnew);
          s[0][r] = p0; s[1][r] = p1;
          float rs = p0 + p1;
          rs += __shfl_xor(rs, 1);
          rs += __shfl_xor(rs, 2);
          rs += __shfl_xor(rs, 4);
          rs += __shfl_xor(rs, 8);
          corr[r] = __expf(mrow[r] - mnew);
          lrow[r] = lrow[r] * corr[r] + rs;
          mrow[r] = mnew;
        }
        #pragma unroll
        for (int j = 0; j < 8; ++j)
          #pragma unroll
          for (int r = 0; r < 8; ++r)
            accO[j][r] = accO[j][r] * corr[r];

        // P (C layout) -> LDS -> A fragment
        _Float16* pw = &p_lds[wave][0];
        #pragma unroll
        for (int sub = 0; sub < 2; ++sub)
          #pragma unroll
          for (int r = 0; r < 8; ++r)
            pw[(r + 8 * half) * PPAD + sub * 16 + l16] = (_Float16)s[sub][r];
        __asm__ volatile("s_wait_dscnt 0" ::: "memory");
        const _Float16* prow = &pw[l16 * PPAD];
        v16h pf = cat8(*(const v8h*)(prow + half * 8),
                       *(const v8h*)(prow + 16 + half * 8));

        auto loadV16 = [&](int j) {
          const _Float16* vrow = &vt_lds[buf][(j * 16 + l16) * VPAD];
          return cat8(*(const v8h*)(vrow + kbase + half * 8),
                      *(const v8h*)(vrow + kbase + 16 + half * 8));
        };
        v16h vf = loadV16(0);
        #pragma unroll
        for (int j = 0; j < 8; ++j) {
          v16h vnx = (j < 7) ? loadV16(j + 1) : vf;
          accO[j] = __builtin_amdgcn_wmma_f32_16x16x32_f16(
              false, pf, false, vf, (short)0, accO[j], false, false);
          vf = vnx;
        }
      }
    }

    __syncthreads();                           // buf^1 readers (kc-1) done
    if (pre) {
      commitChunk(buf ^ 1);
#if HAVE_TDM
      if (wave == 0) __builtin_amdgcn_s_wait_tensorcnt(0);
#endif
    }
    __syncthreads();                           // buf^1 ready for kc+1
  }

  #pragma unroll
  for (int r = 0; r < 8; ++r) {
    float inv = 1.0f / lrow[r];
    int row = qrow0 + r + 8 * half;
    #pragma unroll
    for (int j = 0; j < 8; ++j)
      out[boff + (size_t)row * HS + j * 16 + l16] = accO[j][r] * inv;
  }
}

// ---------------------------------------------------------------------------
extern "C" void kernel_launch(void* const* d_in, const int* in_sizes, int n_in,
                              void* d_out, int out_size, void* d_ws, size_t ws_size,
                              hipStream_t stream) {
  const float* x  = (const float*)d_in[0];
  const float* Wq = (const float*)d_in[1];
  const float* Wk = (const float*)d_in[2];
  const float* Wv = (const float*)d_in[3];

  const size_t n = (size_t)BATCH * TSEQ * HS;
  _Float16* q = (_Float16*)d_ws;
  _Float16* k = q + n;
  _Float16* v = k + n;

  proj_kernel<<<dim3(128, 3), 256, 0, stream>>>(x, Wq, Wk, Wv, q, k, v);
  attn_kernel<<<dim3(32, BATCH), 256, 0, stream>>>(q, k, v, (float*)d_out);
}
